// VM_candidate_model_49160195670366
// MI455X (gfx1250) — compile-verified, matrix operand order checked by reading, not run on hardware
//
#include <hip/hip_runtime.h>
#include <hip/hip_bf16.h>
#include <cmath>

// ---------------- problem constants ----------------
#define BATCH   4
#define NPM_    200
#define NVM_    1500
#define S_TOT   1702          // NPM+NVM+2
#define DMODEL  512
#define HEADS   8
#define DHEAD   64
#define DFF_    2048
#define NLAY    3
#define MROWS   (BATCH * S_TOT)   // 6808
#define MPAD    6912              // MROWS padded to 128-row GEMM tiles
#define PADR    (MPAD - MROWS)    // 104 pad rows
#define NREL    (NPM_ + NVM_)     // 1700
#define SPAD    1728              // S padded to 32-token tiles (54 * 32)

// ---------------- CDNA5 WMMA types ----------------
typedef __bf16        v16bf __attribute__((ext_vector_type(16)));
typedef float         v8f   __attribute__((ext_vector_type(8)));
typedef unsigned int  v4u   __attribute__((ext_vector_type(4)));

__device__ __forceinline__ v8f wmma_bf16(v16bf a, v16bf b, v8f c) {
  // D = A(16x32 bf16) * B(32x16 bf16) + C(16x16 f32)
  return __builtin_amdgcn_wmma_f32_16x16x32_bf16(
      /*neg_a=*/false, a, /*neg_b=*/false, b,
      /*c_mod=*/(short)0, c, /*reuse_a=*/false, /*reuse_b=*/false);
}

__device__ __forceinline__ unsigned short f2bf(float f) {
  unsigned u = __builtin_bit_cast(unsigned, f);
  unsigned r = (u + 0x7fffu + ((u >> 16) & 1u)) >> 16;   // round-to-nearest-even
  return (unsigned short)r;
}

// async 16B/lane copy global -> LDS (ASYNCcnt-tracked, no VGPR round trip).
// Flat LDS addresses carry the LDS byte offset in addr[31:0] (ISA 10.2).
__device__ __forceinline__ void async_g2l_b128(const void* gptr, void* lptr) {
  unsigned lo = (unsigned)(size_t)lptr;
  asm volatile("global_load_async_to_lds_b128 %0, %1, off"
               :: "v"(lo), "v"(gptr) : "memory");
}
__device__ __forceinline__ void wait_async0() {
  asm volatile("s_wait_asynccnt 0x0" ::: "memory");
}

union FragU { v4u q[2]; v16bf v; };

// A/B fragment loader from a row-major [rows][32] bf16 buffer (LDS or global).
// CDNA5 16-bit A layout: lane l holds row (l&15); element e<8 -> K=kb+e,
// e>=8 -> K=16+kb+e-8, with kb=(l>>4)*8.  B layout is the mirrored form
// (lane -> N column), so the same index math applies with row := N.
__device__ __forceinline__ v16bf load_frag_rm32(const unsigned short* buf, int row) {
  int lane  = threadIdx.x & 31;
  int kb8   = (lane >> 4);                 // kbase/8 : 0 or 1
  const v4u* p = (const v4u*)(buf + row * 32);
  FragU u;
  u.q[0] = p[kb8];
  u.q[1] = p[kb8 + 2];
  return u.v;
}

// Same fragment pattern straight from global memory: `base` points at this
// lane's row (of >=32 contiguous K elements); kOff selects the 32-K window.
__device__ __forceinline__ v16bf load_frag_gl(const unsigned short* base, int kOff, bool valid) {
  FragU u;
  if (valid) {
    int lane = threadIdx.x & 31;
    int kb8  = (lane >> 4);
    const v4u* p = (const v4u*)(base + kOff);
    u.q[0] = p[kb8];
    u.q[1] = p[kb8 + 2];
  } else {
    v4u z = {0u, 0u, 0u, 0u};
    u.q[0] = z; u.q[1] = z;
  }
  return u.v;
}

// ---------------- weight convert + transpose (fp32 [K][N] -> bf16 [N][K]) ----
__global__ void wconv_kernel(const float* __restrict__ W,
                             unsigned short* __restrict__ WT, int K, int N) {
  int idx = blockIdx.x * blockDim.x + threadIdx.x;
  if (idx >= K * N) return;
  int n = idx / K, k = idx % K;
  WT[idx] = f2bf(W[(size_t)k * N + n]);
}

// ---------------- V transpose: qkv V-slice -> vT[b][h][dim][token] ----------
__global__ void vtrans_kernel(const unsigned short* __restrict__ qkv,
                              unsigned short* __restrict__ vT) {
  int idx = blockIdx.x * blockDim.x + threadIdx.x;
  if (idx >= BATCH * HEADS * DHEAD * SPAD) return;
  int s  = idx % SPAD;
  int t  = idx / SPAD;
  int d  = t % DHEAD;
  int t2 = t / DHEAD;
  int h  = t2 % HEADS;
  int b  = t2 / HEADS;
  unsigned short v = 0;
  if (s < S_TOT)
    v = qkv[(size_t)(b * S_TOT + s) * 1536 + 1024 + h * DHEAD + d];
  vT[idx] = v;
}

// ---------------- embedding (K=16, negligible FLOPs -> scalar) --------------
__global__ void embed_kernel(const float* __restrict__ vm_states,
                             const float* __restrict__ num_step,
                             const float* __restrict__ pm_states,
                             const float* __restrict__ pm_W, const float* __restrict__ pm_b,
                             const float* __restrict__ vm_W, const float* __restrict__ vm_b,
                             float* __restrict__ x) {
  int idx = blockIdx.x * blockDim.x + threadIdx.x;
  if (idx >= MROWS * DMODEL) return;
  int d   = idx & (DMODEL - 1);
  int row = idx >> 9;
  int b = row / S_TOT, s = row % S_TOT;
  float v;
  if (s == 0) {
    v = num_step[b];
  } else if (s <= NPM_) {
    const float* st = pm_states + ((size_t)b * NPM_ + (s - 1)) * 16;
    float acc = pm_b[d];
    #pragma unroll
    for (int k = 0; k < 16; k++) acc += st[k] * pm_W[k * DMODEL + d];
    v = acc;
  } else if (s <= NPM_ + NVM_) {
    const float* st = vm_states + ((size_t)b * NVM_ + (s - 1 - NPM_)) * 16;
    float acc = vm_b[d];
    #pragma unroll
    for (int k = 0; k < 16; k++) acc += st[k] * vm_W[k * DMODEL + d];
    v = acc;
  } else {
    v = -1.0f;
  }
  x[idx] = v;
}

// ---------------- LayerNorm -> bf16 -----------------------------------------
__global__ void ln_kernel(const float* __restrict__ x,
                          const float* __restrict__ sc, const float* __restrict__ bi,
                          unsigned short* __restrict__ h) {
  int row  = blockIdx.x;
  int tid  = threadIdx.x;               // 256 threads, 2 elems each
  const float* xr = x + (size_t)row * DMODEL;
  float v0 = xr[tid], v1 = xr[tid + 256];
  float sum = v0 + v1;
  for (int m = 1; m < 32; m <<= 1) sum += __shfl_xor(sum, m);
  __shared__ float red[8];
  __shared__ float red2[8];
  int wave = tid >> 5, lane = tid & 31;
  if (lane == 0) red[wave] = sum;
  __syncthreads();
  float tot = 0.f;
  #pragma unroll
  for (int i = 0; i < 8; i++) tot += red[i];
  float mu = tot * (1.0f / DMODEL);
  float d0 = v0 - mu, d1 = v1 - mu;
  float vs = d0 * d0 + d1 * d1;
  for (int m = 1; m < 32; m <<= 1) vs += __shfl_xor(vs, m);
  if (lane == 0) red2[wave] = vs;
  __syncthreads();
  float tv = 0.f;
  #pragma unroll
  for (int i = 0; i < 8; i++) tv += red2[i];
  float inv = rsqrtf(tv * (1.0f / DMODEL) + 1e-5f);
  h[(size_t)row * DMODEL + tid]       = f2bf(d0 * inv * sc[tid]       + bi[tid]);
  h[(size_t)row * DMODEL + tid + 256] = f2bf(d1 * inv * sc[tid + 256] + bi[tid + 256]);
}

// ---------------- bf16 WMMA GEMM: C = A[MPADxK] * W^T[NxK]^T + bias ---------
// 128x64 tile, 128 threads (4 waves); each wave owns 32 rows x 64 cols:
// 8 WMMA per K-step.  Tiles staged with async global->LDS copies (ASYNCcnt);
// no guards anywhere: M==MPAD (buffers padded), N%64==0, K%32==0, EXEC all-1s.
template <bool GELU, bool RESID>
__global__ void gemm_bf16_kernel(const unsigned short* __restrict__ A,
                                 const unsigned short* __restrict__ WT,
                                 const float* __restrict__ bias,
                                 float* __restrict__ Cres,          // fp32 +=, if RESID
                                 unsigned short* __restrict__ Cb,   // bf16 out, if !RESID
                                 int N, int K) {
  __shared__ __align__(16) unsigned short As[128 * 32];
  __shared__ __align__(16) unsigned short Bs[64 * 32];
  int tid  = threadIdx.x;
  int wave = tid >> 5;
  int lane = tid & 31;
  int rowBase = blockIdx.x * 128;
  int colBase = blockIdx.y * 64;

  v8f zero = {0.f, 0.f, 0.f, 0.f, 0.f, 0.f, 0.f, 0.f};
  v8f acc[2][4] = {{zero, zero, zero, zero}, {zero, zero, zero, zero}};

  for (int k0 = 0; k0 < K; k0 += 32) {
    #pragma unroll
    for (int t = 0; t < 4; t++) {       // A tile: 128 rows x 32 K = 512 x 16B
      int idx = tid + t * 128;
      int r   = idx >> 2;
      int c16 = idx & 3;
      async_g2l_b128(A + (size_t)(rowBase + r) * K + k0 + c16 * 8, As + idx * 8);
    }
    #pragma unroll
    for (int t = 0; t < 2; t++) {       // B tile: 64 rows x 32 K = 256 x 16B
      int idx = tid + t * 128;
      int r   = idx >> 2;
      int c16 = idx & 3;
      async_g2l_b128(WT + (size_t)(colBase + r) * K + k0 + c16 * 8, Bs + idx * 8);
    }
    wait_async0();                      // own wave's async copies done
    __syncthreads();                    // all waves' copies done
    v16bf af0 = load_frag_rm32(As, wave * 32 +      (lane & 15));
    v16bf af1 = load_frag_rm32(As, wave * 32 + 16 + (lane & 15));
    #pragma unroll
    for (int nc = 0; nc < 4; nc++) {
      v16bf bf = load_frag_rm32(Bs, nc * 16 + (lane & 15));
      acc[0][nc] = wmma_bf16(af0, bf, acc[0][nc]);
      acc[1][nc] = wmma_bf16(af1, bf, acc[1][nc]);
    }
    __syncthreads();
  }

  // epilogue: C layout -> lane holds col (lane&15), rows r + (lane>>4)*8
  int hf = lane >> 4;
  int n0 = lane & 15;
  #pragma unroll
  for (int mi = 0; mi < 2; mi++) {
    #pragma unroll
    for (int nc = 0; nc < 4; nc++) {
      int col = colBase + nc * 16 + n0;
      float bv = bias[col];
      #pragma unroll
      for (int r = 0; r < 8; r++) {
        int row = rowBase + wave * 32 + mi * 16 + r + hf * 8;
        float v = acc[mi][nc][r] + bv;
        if (GELU) v = 0.5f * v * (1.0f + erff(v * 0.70710678118654752f));
        if (RESID) Cres[(size_t)row * N + col] += v;
        else       Cb[(size_t)row * N + col] = f2bf(v);
      }
    }
  }
}

// ---------------- flash attention: one wave per (b, h, 16-row q tile) -------
__global__ void attn_kernel(const unsigned short* __restrict__ qkv, // rows x 1536
                            const unsigned short* __restrict__ vT,  // B*H*64 x SPAD
                            const int* __restrict__ rel,            // B x 1700
                            const unsigned char* __restrict__ padm, // B x NVM (bool)
                            unsigned short* __restrict__ o) {       // rows x 512
  __shared__ __align__(16) unsigned short Pl[16 * 32];
  int qt = blockIdx.x;
  int hh = blockIdx.y;
  int b  = blockIdx.z;
  int lane = threadIdx.x;
  int n0 = lane & 15;
  int hf = lane >> 4;
  const float scale = 0.125f;   // DH^-0.5

  // Q fragments (A layout: lane row = qt*16 + (lane&15))
  int qlA = qt * 16 + n0;
  bool qAv = (qlA < S_TOT);
  const unsigned short* qrow =
      qkv + (size_t)(b * S_TOT + (qAv ? qlA : 0)) * 1536 + hh * DHEAD;
  v16bf qa0 = load_frag_gl(qrow, 0,  qAv);
  v16bf qa1 = load_frag_gl(qrow, 32, qAv);

  // this lane's V^T rows (dim = nc*16 + n0), contiguous along tokens
  const unsigned short* vrow[4];
  #pragma unroll
  for (int nc = 0; nc < 4; nc++)
    vrow[nc] = vT + (size_t)((b * HEADS + hh) * DHEAD + nc * 16 + n0) * SPAD;

  // per-output-row (C-layout rows) mask info
  int relq[8]; bool qin[8];
  #pragma unroll
  for (int r = 0; r < 8; r++) {
    int q = qt * 16 + r + hf * 8;
    bool in = (q >= 1) && (q < S_TOT - 1);
    qin[r] = in;
    relq[r] = in ? rel[b * NREL + q - 1] : -1;
  }

  v8f zero = {0.f, 0.f, 0.f, 0.f, 0.f, 0.f, 0.f, 0.f};
  v8f O[4] = {zero, zero, zero, zero};
  float mrow[8], lrow[8];
  #pragma unroll
  for (int r = 0; r < 8; r++) { mrow[r] = -INFINITY; lrow[r] = 0.f; }

  const int ITER = SPAD / 32;               // 54 iterations of 32 k-tokens
  for (int it = 0; it < ITER; it++) {
    int k0 = it * 32;

    float scv[2][8];
    #pragma unroll
    for (int kt = 0; kt < 2; kt++) {
      int ktok = k0 + kt * 16 + n0;          // B layout: lane -> k token
      bool kv = ktok < S_TOT;
      const unsigned short* krow =
          qkv + (size_t)(b * S_TOT + (kv ? ktok : 0)) * 1536 + 512 + hh * DHEAD;
      v16bf kb0 = load_frag_gl(krow, 0,  kv);
      v16bf kb1 = load_frag_gl(krow, 32, kv);
      v8f c = zero;
      c = wmma_bf16(qa0, kb0, c);
      c = wmma_bf16(qa1, kb1, c);
      bool kin  = (ktok >= 1) && (ktok < S_TOT - 1);
      int  relk = kin ? rel[b * NREL + ktok - 1] : -1;
      bool padk = (ktok >= 1 + NPM_) && (ktok < S_TOT - 1) &&
                  (padm[b * NVM_ + ktok - 1 - NPM_] != 0);
      #pragma unroll
      for (int r = 0; r < 8; r++) {
        float v = c[r] * scale;
        bool msk = (qin[r] && kin && (relq[r] != relk)) || padk;
        if (msk) v = -1.0e9f;                // matches reference "neg"
        if (!kv) v = -INFINITY;              // out-of-sequence padding tokens
        scv[kt][r] = v;
      }
    }

    // streaming softmax (row reductions across the 16-lane half-wave)
    #pragma unroll
    for (int r = 0; r < 8; r++) {
      float mx = fmaxf(scv[0][r], scv[1][r]);
      for (int m = 1; m < 16; m <<= 1) mx = fmaxf(mx, __shfl_xor(mx, m));
      float mnew = fmaxf(mrow[r], mx);
      float corr = expf(mrow[r] - mnew);
      float p0 = expf(scv[0][r] - mnew);
      float p1 = expf(scv[1][r] - mnew);
      float rs = p0 + p1;
      for (int m = 1; m < 16; m <<= 1) rs += __shfl_xor(rs, m);
      lrow[r] = lrow[r] * corr + rs;
      mrow[r] = mnew;
      #pragma unroll
      for (int nc = 0; nc < 4; nc++) O[nc][r] *= corr;
      // stage P (16x32) into LDS in [row][k] order for A-frag reload
      Pl[(r + hf * 8) * 32 + n0]      = f2bf(p0);
      Pl[(r + hf * 8) * 32 + 16 + n0] = f2bf(p1);
    }
    __syncthreads();

    v16bf pa = load_frag_rm32(Pl, n0);       // P as A matrix (16x32)
    #pragma unroll
    for (int nc = 0; nc < 4; nc++) {         // V^T rows: contiguous B frags
      v16bf bv = load_frag_gl(vrow[nc], k0, true);
      O[nc] = wmma_bf16(pa, bv, O[nc]);
    }
    __syncthreads();
  }

  // normalize and store per-head output as bf16
  #pragma unroll
  for (int nc = 0; nc < 4; nc++) {
    #pragma unroll
    for (int r = 0; r < 8; r++) {
      int q = qt * 16 + r + hf * 8;
      if (q < S_TOT) {
        float v = O[nc][r] / lrow[r];
        o[(size_t)(b * S_TOT + q) * DMODEL + hh * DHEAD + nc * 16 + n0] = f2bf(v);
      }
    }
  }
}

// ---------------- output projections ----------------------------------------
__global__ void outproj_kernel(const float* __restrict__ x,
                               const float* __restrict__ outW, const float* __restrict__ outB,
                               const float* __restrict__ crW,  const float* __restrict__ crB,
                               float* __restrict__ out) {
  int gw   = (blockIdx.x * blockDim.x + threadIdx.x) >> 5;
  int lane = threadIdx.x & 31;
  const int total = BATCH * (NVM_ + 1);
  if (gw >= total) return;
  int b = gw / (NVM_ + 1);
  int j = gw % (NVM_ + 1);
  const float* w; float bias; size_t row; int oidx;
  if (j < NVM_) { row = (size_t)b * S_TOT + 1 + NPM_ + j; w = outW; bias = outB[0]; oidx = b * NVM_ + j; }
  else          { row = (size_t)b * S_TOT + S_TOT - 1;    w = crW;  bias = crB[0];  oidx = BATCH * NVM_ + b; }
  const float* xr = x + row * DMODEL;
  float acc = 0.f;
  for (int k = lane; k < DMODEL; k += 32) acc += xr[k] * w[k];
  for (int m = 1; m < 32; m <<= 1) acc += __shfl_xor(acc, m);
  if (lane == 0) out[oidx] = acc + bias;
}

// ---------------- driver -----------------------------------------------------
extern "C" void kernel_launch(void* const* d_in, const int* in_sizes, int n_in,
                              void* d_out, int out_size, void* d_ws, size_t ws_size,
                              hipStream_t stream) {
  const float* vm_states = (const float*)d_in[0];
  const float* num_step  = (const float*)d_in[1];
  const float* pm_states = (const float*)d_in[2];
  const int*   rel       = (const int*)d_in[3];
  const unsigned char* padm = (const unsigned char*)d_in[4];
  const float* pm_W  = (const float*)d_in[5];
  const float* pm_b  = (const float*)d_in[6];
  const float* vm_W  = (const float*)d_in[7];
  const float* vm_b  = (const float*)d_in[8];
  const float* ln1_s = (const float*)d_in[9];
  const float* ln1_b = (const float*)d_in[10];
  const float* Wqkv  = (const float*)d_in[11];
  const float* bqkv  = (const float*)d_in[12];
  const float* Wo    = (const float*)d_in[13];
  const float* bo    = (const float*)d_in[14];
  const float* ln2_s = (const float*)d_in[15];
  const float* ln2_b = (const float*)d_in[16];
  const float* W1    = (const float*)d_in[17];
  const float* b1    = (const float*)d_in[18];
  const float* W2    = (const float*)d_in[19];
  const float* b2    = (const float*)d_in[20];
  const float* out_W = (const float*)d_in[21];
  const float* out_b = (const float*)d_in[22];
  const float* cr_W  = (const float*)d_in[23];
  const float* cr_b  = (const float*)d_in[24];
  float* out = (float*)d_out;

  // workspace layout (activation buffers padded to MPAD rows)
  char* w = (char*)d_ws;
  float* x = (float*)w;                      w += (size_t)MPAD * DMODEL * 4;
  unsigned short* h    = (unsigned short*)w; w += (size_t)MPAD * DMODEL * 2;
  unsigned short* qkvb = (unsigned short*)w; w += (size_t)MPAD * 3 * DMODEL * 2;
  unsigned short* obuf = (unsigned short*)w; w += (size_t)MPAD * DMODEL * 2;
  unsigned short* ffn  = (unsigned short*)w; w += (size_t)MPAD * DFF_ * 2;
  unsigned short* vTb  = (unsigned short*)w; w += (size_t)BATCH * HEADS * DHEAD * SPAD * 2;
  unsigned short* WqkvT = (unsigned short*)w; w += (size_t)NLAY * 3 * DMODEL * DMODEL * 2;
  unsigned short* WoT   = (unsigned short*)w; w += (size_t)NLAY * DMODEL * DMODEL * 2;
  unsigned short* W1T   = (unsigned short*)w; w += (size_t)NLAY * DMODEL * DFF_ * 2;
  unsigned short* W2T   = (unsigned short*)w; w += (size_t)NLAY * DFF_ * DMODEL * 2;

  const int MT128 = MPAD / 128;            // 54 row tiles (exact)
  const int QT    = (S_TOT + 15) / 16;     // 107 q tiles (attention)

  // zero the pad strips so unguarded GEMMs stay deterministic across replays
  hipMemsetAsync(x    + (size_t)MROWS * DMODEL,     0, (size_t)PADR * DMODEL * 4, stream);
  hipMemsetAsync(h    + (size_t)MROWS * DMODEL,     0, (size_t)PADR * DMODEL * 2, stream);
  hipMemsetAsync(qkvb + (size_t)MROWS * 3 * DMODEL, 0, (size_t)PADR * 3 * DMODEL * 2, stream);
  hipMemsetAsync(obuf + (size_t)MROWS * DMODEL,     0, (size_t)PADR * DMODEL * 2, stream);
  hipMemsetAsync(ffn  + (size_t)MROWS * DFF_,       0, (size_t)PADR * DFF_ * 2, stream);

  // bf16 transposed weight copies (L2-resident afterwards)
  for (int i = 0; i < NLAY; i++) {
    {
      int kn = DMODEL * 3 * DMODEL;
      wconv_kernel<<<(kn + 255) / 256, 256, 0, stream>>>(
          Wqkv + (size_t)i * kn, WqkvT + (size_t)i * kn, DMODEL, 3 * DMODEL);
    }
    {
      int kn = DMODEL * DMODEL;
      wconv_kernel<<<(kn + 255) / 256, 256, 0, stream>>>(
          Wo + (size_t)i * kn, WoT + (size_t)i * kn, DMODEL, DMODEL);
    }
    {
      int kn = DMODEL * DFF_;
      wconv_kernel<<<(kn + 255) / 256, 256, 0, stream>>>(
          W1 + (size_t)i * kn, W1T + (size_t)i * kn, DMODEL, DFF_);
      wconv_kernel<<<(kn + 255) / 256, 256, 0, stream>>>(
          W2 + (size_t)i * kn, W2T + (size_t)i * kn, DFF_, DMODEL);
    }
  }

  embed_kernel<<<(MROWS * DMODEL + 255) / 256, 256, 0, stream>>>(
      vm_states, num_step, pm_states, pm_W, pm_b, vm_W, vm_b, x);

  for (int i = 0; i < NLAY; i++) {
    ln_kernel<<<MROWS, 256, 0, stream>>>(x, ln1_s + i * DMODEL, ln1_b + i * DMODEL, h);
    gemm_bf16_kernel<false, false><<<dim3(MT128, 24), 128, 0, stream>>>(
        h, WqkvT + (size_t)i * 3 * DMODEL * DMODEL, bqkv + (size_t)i * 3 * DMODEL,
        nullptr, qkvb, 3 * DMODEL, DMODEL);
    {
      int n = BATCH * HEADS * DHEAD * SPAD;
      vtrans_kernel<<<(n + 255) / 256, 256, 0, stream>>>(qkvb, vTb);
    }
    attn_kernel<<<dim3(QT, HEADS, BATCH), 32, 0, stream>>>(qkvb, vTb, rel, padm, obuf);
    gemm_bf16_kernel<false, true><<<dim3(MT128, 8), 128, 0, stream>>>(
        obuf, WoT + (size_t)i * DMODEL * DMODEL, bo + (size_t)i * DMODEL,
        x, nullptr, DMODEL, DMODEL);
    ln_kernel<<<MROWS, 256, 0, stream>>>(x, ln2_s + i * DMODEL, ln2_b + i * DMODEL, h);
    gemm_bf16_kernel<true, false><<<dim3(MT128, 32), 128, 0, stream>>>(
        h, W1T + (size_t)i * DMODEL * DFF_, b1 + (size_t)i * DFF_,
        nullptr, ffn, DFF_, DMODEL);
    gemm_bf16_kernel<false, true><<<dim3(MT128, 8), 128, 0, stream>>>(
        ffn, W2T + (size_t)i * DFF_ * DMODEL, b2 + (size_t)i * DMODEL,
        x, nullptr, DMODEL, DFF_);
  }

  {
    int waves  = BATCH * (NVM_ + 1);
    int blocks = (waves + 3) / 4;       // 4 waves / 128-thread block
    outproj_kernel<<<blocks, 128, 0, stream>>>(x, out_W, out_b, cr_W, cr_b, out);
  }
}
